// MergeDNALayer_90288802496787
// MI455X (gfx1250) — compile-verified, matrix operand order checked by reading, not run on hardware
//
#include <hip/hip_runtime.h>
#include <hip/hip_bf16.h>

// ---------------------------------------------------------------------------
// MergeDNALayer for MI455X (gfx1250, wave32, WMMA)
// B=4, N=8192, C=512, Cm=128, r=2048
// ---------------------------------------------------------------------------

typedef __attribute__((ext_vector_type(16))) __bf16 v16bf;
typedef __attribute__((ext_vector_type(8)))  float  v8f;

#define BB   4
#define NN   8192
#define CC   512
#define CM   128
#define RR   2048
#define NA   4096          // N/2 token pairs
#define NKEEP 6144         // N - r

union BfPack { uint4 u; __bf16 h[8]; };

// ---------------------------------------------------------------------------
// K1: metric = normalize(x @ W.T + b), stored as bf16 bits.
// One wave per row (row = b*N + n). 8 waves / 256-thread block.
// ---------------------------------------------------------------------------
__global__ void metric_kernel(const float* __restrict__ x,
                              const float* __restrict__ W,
                              const float* __restrict__ bias,
                              unsigned short* __restrict__ metric) {
    __shared__ float xs[8][CC];
    const int lane = threadIdx.x & 31;
    const int wave = threadIdx.x >> 5;
    const size_t row = (size_t)blockIdx.x * 8 + wave;      // < B*N
    const float* xr = x + row * CC;

    // cooperative row load (float4 per lane, 4 chunks)
#pragma unroll
    for (int j = 0; j < 4; ++j)
        *(float4*)&xs[wave][lane * 4 + j * 128] = *(const float4*)&xr[lane * 4 + j * 128];
    __syncthreads();

    float acc[4];
#pragma unroll
    for (int j = 0; j < 4; ++j) acc[j] = bias[lane + 32 * j];

#pragma unroll 4
    for (int c = 0; c < CC; ++c) {
        float xv = xs[wave][c];
#pragma unroll
        for (int j = 0; j < 4; ++j)
            acc[j] += xv * W[(size_t)(lane + 32 * j) * CC + c];
    }

    float ss = 0.f;
#pragma unroll
    for (int j = 0; j < 4; ++j) ss += acc[j] * acc[j];
#pragma unroll
    for (int off = 1; off < 32; off <<= 1) ss += __shfl_xor(ss, off, 32);
    const float norm = sqrtf(ss);

    unsigned short* out = metric + row * CM;
#pragma unroll
    for (int j = 0; j < 4; ++j) {
        float v = acc[j] / norm;
        unsigned u = __float_as_uint(v);
        u += 0x7fffu + ((u >> 16) & 1u);          // round-to-nearest-even bf16
        out[lane + 32 * j] = (unsigned short)(u >> 16);
    }
}

// ---------------------------------------------------------------------------
// K2: fused scores + row max/argmax via v_wmma_f32_16x16x32_bf16.
// Each wave owns TWO 16-row A tiles (32 even tokens) and iterates all 256
// 16-column B tiles (odd tokens); K = 128 = 4 x 32.  B fragments are loaded
// once per tile and reused for both accumulators -> 8 WMMAs per load set.
// Running max/argmax kept in the 8-VGPR f32 accumulator layout; final
// cross-lane reduce over the 16-lane halves.
// ---------------------------------------------------------------------------
__device__ inline v16bf load_frag(const unsigned short* __restrict__ rowPtr,
                                  int kk, int lane) {
    // 16-bit A/B fragment layout (ISA 7.12.2): lane holds row (lane&15),
    // kbase = (lane<16 ? 0 : 8); elems 0..7 -> K kbase..kbase+7,
    // elems 8..15 -> K kbase+16..kbase+23, all offset by kk*32.
    const int kbase = kk * 32 + ((lane & 16) ? 8 : 0);
    BfPack lo, hi;
    lo.u = *(const uint4*)(rowPtr + kbase);
    hi.u = *(const uint4*)(rowPtr + kbase + 16);
    v16bf f;
#pragma unroll
    for (int j = 0; j < 8; ++j) { f[j] = lo.h[j]; f[8 + j] = hi.h[j]; }
    return f;
}

__global__ void scores_rowmax_kernel(const unsigned short* __restrict__ metric,
                                     float* __restrict__ values,
                                     int* __restrict__ node) {
    const int lane  = threadIdx.x & 31;
    const int wave  = threadIdx.x >> 5;
    const int pair  = blockIdx.x * 4 + wave;     // 0..511 (2 M-tiles each)
    const int b     = pair >> 7;                 // /128
    const int mb0   = (pair & 127) << 5;         // first tile rows  [mb0,   mb0+16)
    const int mb1   = mb0 + 16;                  // second tile rows [mb0+16,mb0+32)
    const unsigned short* mbase = metric + (size_t)b * NN * CM;

    // preload A fragments for both tiles: rows = even tokens 2*(mb + (lane&15))
    const unsigned short* aptr0 = mbase + (size_t)(2 * (mb0 + (lane & 15))) * CM;
    const unsigned short* aptr1 = mbase + (size_t)(2 * (mb1 + (lane & 15))) * CM;
    v16bf afrag0[4], afrag1[4];
#pragma unroll
    for (int kk = 0; kk < 4; ++kk) {
        afrag0[kk] = load_frag(aptr0, kk, lane);
        afrag1[kk] = load_frag(aptr1, kk, lane);
    }

    float best0[8], best1[8];
    int   bidx0[8], bidx1[8];
#pragma unroll
    for (int m = 0; m < 8; ++m) {
        best0[m] = -3.4e38f; bidx0[m] = 0;
        best1[m] = -3.4e38f; bidx1[m] = 0;
    }

    for (int ct = 0; ct < 256; ++ct) {
        // B columns = odd tokens 2*(ct*16 + (lane&15)) + 1
        const unsigned short* bptr =
            mbase + (size_t)(2 * (ct * 16 + (lane & 15)) + 1) * CM;
        v16bf bfrag[4];
#pragma unroll
        for (int kk = 0; kk < 4; ++kk) bfrag[kk] = load_frag(bptr, kk, lane);

        v8f acc0 = {}, acc1 = {};
#pragma unroll
        for (int kk = 0; kk < 4; ++kk)
            acc0 = __builtin_amdgcn_wmma_f32_16x16x32_bf16(
                false, afrag0[kk], false, bfrag[kk], (short)0, acc0, false, false);
#pragma unroll
        for (int kk = 0; kk < 4; ++kk)
            acc1 = __builtin_amdgcn_wmma_f32_16x16x32_bf16(
                false, afrag1[kk], false, bfrag[kk], (short)0, acc1, false, false);

        const int cidx = ct * 16 + (lane & 15);
#pragma unroll
        for (int m = 0; m < 8; ++m) {
            float v0 = acc0[m];
            if (v0 > best0[m]) { best0[m] = v0; bidx0[m] = cidx; }  // keep first max
            float v1 = acc1[m];
            if (v1 > best1[m]) { best1[m] = v1; bidx1[m] = cidx; }
        }
    }

    // reduce across the 16 lanes of each half (masks < 16 stay in-group)
#pragma unroll
    for (int off = 1; off < 16; off <<= 1) {
#pragma unroll
        for (int m = 0; m < 8; ++m) {
            float ov0 = __shfl_xor(best0[m], off, 32);
            int   oi0 = __shfl_xor(bidx0[m], off, 32);
            if (ov0 > best0[m] || (ov0 == best0[m] && oi0 < bidx0[m])) {
                best0[m] = ov0; bidx0[m] = oi0;
            }
            float ov1 = __shfl_xor(best1[m], off, 32);
            int   oi1 = __shfl_xor(bidx1[m], off, 32);
            if (ov1 > best1[m] || (ov1 == best1[m] && oi1 < bidx1[m])) {
                best1[m] = ov1; bidx1[m] = oi1;
            }
        }
    }

    if ((lane & 15) == 0) {
        const int half = lane >> 4;              // lanes 0-15 -> M 0..7, 16-31 -> M 8..15
#pragma unroll
        for (int m = 0; m < 8; ++m) {
            const int row0 = mb0 + m + half * 8;
            values[b * NA + row0] = best0[m];
            node  [b * NA + row0] = bidx0[m];
            const int row1 = mb1 + m + half * 8;
            values[b * NA + row1] = best1[m];
            node  [b * NA + row1] = bidx1[m];
        }
    }
}

// ---------------------------------------------------------------------------
// K3: top-k membership by rank counting (exactly matches jax top_k tie rule:
// element selected iff  #{j : v[j]>v[i] or (v[j]==v[i] and j<i)} < r).
// One 1024-thread block per batch, values staged in LDS.
// ---------------------------------------------------------------------------
__global__ void topk_select_kernel(const float* __restrict__ values,
                                   int* __restrict__ sel) {
    __shared__ float sv[NA];
    const int b = blockIdx.x;
    const int t = threadIdx.x;
    for (int k = t; k < NA; k += 1024) sv[k] = values[b * NA + k];
    __syncthreads();
#pragma unroll
    for (int j = 0; j < 4; ++j) {
        const int i = t + 1024 * j;
        const float vi = sv[i];
        int rank = 0;
        for (int k = 0; k < NA; ++k) {
            const float vk = sv[k];
            rank += (vk > vi) || (vk == vi && k < i);
        }
        sel[b * NA + i] = (rank < RR) ? 1 : 0;
    }
}

// ---------------------------------------------------------------------------
// K4: new_idx = cumsum(mask)-1 over 8192 tokens per batch.
// mask[n] = !(n odd && sel[n/2]).  1024 threads, 8 elems/thread,
// Hillis-Steele scan of the 1024 chunk totals in LDS.
// ---------------------------------------------------------------------------
__global__ void prefix_kernel(const int* __restrict__ sel,
                              int* __restrict__ newidx) {
    __shared__ int tmp[1024];
    const int b = blockIdx.x;
    const int t = threadIdx.x;
    int loc[8];
    int s = 0;
#pragma unroll
    for (int j = 0; j < 8; ++j) {
        const int n = t * 8 + j;
        const int m = ((n & 1) && sel[b * NA + (n >> 1)]) ? 0 : 1;
        s += m;
        loc[j] = s;
    }
    tmp[t] = s;
    __syncthreads();
    for (int off = 1; off < 1024; off <<= 1) {
        const int v = (t >= off) ? tmp[t - off] : 0;
        __syncthreads();
        tmp[t] += v;
        __syncthreads();
    }
    const int prev = (t > 0) ? tmp[t - 1] : 0;
#pragma unroll
    for (int j = 0; j < 8; ++j)
        newidx[b * NN + t * 8 + j] = prev + loc[j] - 1;
}

// ---------------------------------------------------------------------------
// K5: final merge + compact + ownership. One wave per token row.
// Removed rows (odd & selected): write ownership only.
// Even rows: deterministic ascending-order gather of merged odd partners,
// then divide by count (matches reference x_out/counts).
// ---------------------------------------------------------------------------
__global__ void merge_output_kernel(const float* __restrict__ x,
                                    const int* __restrict__ sel,
                                    const int* __restrict__ node,
                                    const int* __restrict__ newidx,
                                    float* __restrict__ xout,
                                    int* __restrict__ own) {
    const int lane = threadIdx.x & 31;
    const int wave = threadIdx.x >> 5;
    const int idx  = blockIdx.x * 8 + wave;      // 0..B*N-1
    const int b    = idx >> 13;                  // /8192
    const int n    = idx & (NN - 1);
    const int* selb  = sel    + b * NA;
    const int* nodeb = node   + b * NA;
    const int* nib   = newidx + b * NN;

    if ((n & 1) && selb[n >> 1]) {               // removed odd token
        if (lane == 0) own[b * NN + n] = nib[2 * nodeb[n >> 1]];
        return;
    }

    const int p = nib[n];
    if (lane == 0) own[b * NN + n] = p;

    const float* xr = x + ((size_t)b * NN + n) * CC;
    float4 acc[4];
#pragma unroll
    for (int q = 0; q < 4; ++q) acc[q] = *(const float4*)&xr[lane * 16 + q * 4];

    if ((n & 1) == 0) {
        const int ap = n >> 1;
        int cnt = 1;
        for (int i = 0; i < NA; ++i) {           // wave-uniform scan, ascending
            if (selb[i] && nodeb[i] == ap) {
                const float* sr = x + ((size_t)b * NN + 2 * i + 1) * CC;
#pragma unroll
                for (int q = 0; q < 4; ++q) {
                    const float4 sv = *(const float4*)&sr[lane * 16 + q * 4];
                    acc[q].x += sv.x; acc[q].y += sv.y;
                    acc[q].z += sv.z; acc[q].w += sv.w;
                }
                ++cnt;
            }
        }
        const float cf = (float)cnt;
#pragma unroll
        for (int q = 0; q < 4; ++q) {
            acc[q].x /= cf; acc[q].y /= cf; acc[q].z /= cf; acc[q].w /= cf;
        }
    }

    float* orow = xout + ((size_t)b * NKEEP + p) * CC;
#pragma unroll
    for (int q = 0; q < 4; ++q)
        *(float4*)&orow[lane * 16 + q * 4] = acc[q];
}

// ---------------------------------------------------------------------------
extern "C" void kernel_launch(void* const* d_in, const int* in_sizes, int n_in,
                              void* d_out, int out_size, void* d_ws, size_t ws_size,
                              hipStream_t stream) {
    const float* x    = (const float*)d_in[0];   // (B,N,C)
    const float* W    = (const float*)d_in[1];   // (128,512)
    const float* bias = (const float*)d_in[2];   // (128,)

    char* ws = (char*)d_ws;
    unsigned short* metric = (unsigned short*)ws;                 // B*N*128 bf16  = 8 MB
    size_t off = (size_t)BB * NN * CM * sizeof(unsigned short);
    float* values = (float*)(ws + off);  off += (size_t)BB * NA * sizeof(float);
    int*   node   = (int*)  (ws + off);  off += (size_t)BB * NA * sizeof(int);
    int*   sel    = (int*)  (ws + off);  off += (size_t)BB * NA * sizeof(int);
    int*   nidx   = (int*)  (ws + off);  off += (size_t)BB * NN * sizeof(int);

    float* xout = (float*)d_out;                                   // B*6144*512 f32
    int*   own  = (int*)d_out + (size_t)BB * NKEEP * CC;           // B*8192 int32 (raw bits)

    metric_kernel       <<<BB * NN / 8, 256, 0, stream>>>(x, W, bias, metric);
    scores_rowmax_kernel<<<BB * 128 / 4, 128, 0, stream>>>(metric, values, node);
    topk_select_kernel  <<<BB, 1024, 0, stream>>>(values, sel);
    prefix_kernel       <<<BB, 1024, 0, stream>>>(sel, nidx);
    merge_output_kernel <<<BB * NN / 8, 256, 0, stream>>>(x, sel, node, nidx, xout, own);
}
